// GAT_61220463837887
// MI455X (gfx1250) — matrix operand order repaired and checked by reading, not compiled, not run on hardware
//
#include <hip/hip_runtime.h>
#include <stdint.h>

typedef __bf16 bf16_t;
typedef bf16_t v16bf __attribute__((ext_vector_type(16)));
typedef float  v8f   __attribute__((ext_vector_type(8)));

union Frag32 { uint4 u[2]; v16bf v; };

__device__ __forceinline__ unsigned short f2bf(float f) {
  union { float f; unsigned u; } x; x.f = f;
  unsigned r = x.u + 0x7FFFu + ((x.u >> 16) & 1u);  // round-to-nearest-even
  return (unsigned short)(r >> 16);
}

__device__ __forceinline__ v8f zero8() {
  v8f z = {0.f,0.f,0.f,0.f,0.f,0.f,0.f,0.f};
  return z;
}

// CDNA5 async copy: global -> LDS, 16B per lane, tracked by ASYNCcnt.
// lds_off is the 32-bit LDS byte address (low word of the shared-aperture
// generic address), gsrc a per-lane 64-bit global address.
__device__ __forceinline__ void async_copy_b128(unsigned lds_off, const void* gsrc) {
  asm volatile("global_load_async_to_lds_b128 %0, %1, off"
               :: "v"(lds_off), "v"(gsrc)
               : "memory");
}

// ---------------------------------------------------------------------------
// Kernel 1: pack/transpose weights to bf16 (one-time, L2-resident afterwards)
// ---------------------------------------------------------------------------
__global__ void pack_weights(const float* __restrict__ Wq, const float* __restrict__ Wk,
                             const float* __restrict__ Wv, const float* __restrict__ Wo,
                             const float* __restrict__ bq, const float* __restrict__ bk,
                             const float* __restrict__ bv,
                             unsigned short* __restrict__ WcatT,
                             unsigned short* __restrict__ WoT,
                             float* __restrict__ bias_cat) {
  const long total_cat = 1536L * 2304L;
  const long total_o   = 768L * 768L;
  const long n_all     = total_cat + total_o + 2304L;
  for (long idx = (long)blockIdx.x * blockDim.x + threadIdx.x; idx < n_all;
       idx += (long)gridDim.x * blockDim.x) {
    if (idx < total_cat) {
      long k = idx / 2304, n = idx % 2304;      // consecutive threads: coalesced read
      int seg = (int)(n / 768), n0 = (int)(n % 768);
      const float* W = (seg == 0) ? Wq : ((seg == 1) ? Wk : Wv);
      WcatT[n * 1536 + k] = f2bf(W[k * 768 + n0]);
    } else if (idx < total_cat + total_o) {
      long j = idx - total_cat;
      long k = j / 768, n = j % 768;
      WoT[n * 768 + k] = f2bf(Wo[k * 768 + n]);
    } else {
      long j = idx - total_cat - total_o;
      int seg = (int)(j / 768), n0 = (int)(j % 768);
      const float* bs = (seg == 0) ? bq : ((seg == 1) ? bk : bv);
      bias_cat[j] = bs[n0];
    }
  }
}

// ---------------------------------------------------------------------------
// Kernel 2/4: bf16 WMMA GEMM.  acc[M,N] = A[M,K] @ WT[N,K]^T + bias[N]
//   Block: 256 threads (8 wave32), C tile 128(M) x 256(N); wave tile 64x64.
//   A double-buffered through LDS: next slab's global loads issued at the top
//   of each K-step (in flight during the WMMAs), converted + stored to the
//   other LDS buffer at the tail => one barrier per K-step.
//   B fragments from global (L2-resident weights) with one-step register
//   prefetch so WMMAs never wait on a just-issued load.
//   OUT_MODE 0: fp32 -> Cp (row stride N)
//   OUT_MODE 1: QKV split: n<1536 -> bf16 qk[M,1536]; n>=1536 -> transposed
//               bf16 vt[b, h, dh, key]  (b = m>>11, key = m&2047, L = 2048),
//               vectorized: 8 consecutive keys per lane -> one b128 store.
// ---------------------------------------------------------------------------
template <int A_BF16, int OUT_MODE>
__global__ __launch_bounds__(256) void gemm_bf16_wmma(
    const void* __restrict__ Ap, const unsigned short* __restrict__ WT,
    const float* __restrict__ bias, void* __restrict__ Cp,
    unsigned short* __restrict__ vtOut,
    int M, int N, int K) {
  __shared__ __align__(16) unsigned short lds_a[2][128 * 32];  // 2 x 8 KB A tiles

  const int tid  = threadIdx.x;
  const int lane = tid & 31;
  const int wave = tid >> 5;
  const int l15  = lane & 15;
  const int lhi  = lane >> 4;                 // lane half: selects K sub-range
  const int mtile = blockIdx.y * 128;
  const int ntile = blockIdx.x * 256;
  const int msub  = (wave & 1) * 64;
  const int nsub  = (wave >> 1) * 64;

  v8f acc[4][4];
#pragma unroll
  for (int i = 0; i < 4; i++)
#pragma unroll
    for (int j = 0; j < 4; j++) acc[i][j] = zero8();

  const int arow  = tid >> 1;                 // 0..127: row of A tile
  const int kpart = (tid & 1) * 16;           // half of the 32-wide K slab

  // Raw staging registers for the next A slab (global -> reg -> LDS).
  float4 graw[4];
  uint4  gbf[2];

  auto loadA = [&](int k0) {
    if (A_BF16) {
      const uint4* src = (const uint4*)((const unsigned short*)Ap +
                                        (long)(mtile + arow) * K + k0 + kpart);
      gbf[0] = src[0];
      gbf[1] = src[1];
    } else {
      const float* abase = (const float*)Ap + (long)(mtile + arow) * K + k0 + kpart;
      if (k0 + 32 < K) __builtin_prefetch(abase + 32, 0, 1);  // global_prefetch_b8
      const float4* src = (const float4*)abase;
#pragma unroll
      for (int q = 0; q < 4; q++) graw[q] = src[q];
    }
  };
  auto storeA = [&](int buf) {   // convert (if needed) at store time
    uint4* dst = (uint4*)&lds_a[buf][arow * 32 + kpart];
    if (A_BF16) {
      dst[0] = gbf[0];
      dst[1] = gbf[1];
    } else {
      union { unsigned short s[16]; uint4 u[2]; } tmp;
#pragma unroll
      for (int q = 0; q < 4; q++) {
        tmp.s[q * 4 + 0] = f2bf(graw[q].x);
        tmp.s[q * 4 + 1] = f2bf(graw[q].y);
        tmp.s[q * 4 + 2] = f2bf(graw[q].z);
        tmp.s[q * 4 + 3] = f2bf(graw[q].w);
      }
      dst[0] = tmp.u[0];
      dst[1] = tmp.u[1];
    }
  };

  // B fragment loader: lane holds col n = l15, 16 contiguous K at lhi*16
  auto loadB = [&](int k0, Frag32* bf) {
#pragma unroll
    for (int j = 0; j < 4; j++) {
      const int n = ntile + nsub + j * 16 + l15;
      const uint4* w = (const uint4*)(WT + (long)n * K + k0 + lhi * 16);
      bf[j].u[0] = w[0];
      bf[j].u[1] = w[1];
    }
  };

  loadA(0);
  storeA(0);
  Frag32 bcur[4];
  loadB(0, bcur);

  const int nk = K / 32;
  for (int j = 0; j < nk; ++j) {
    if (j + 1 < nk) loadA((j + 1) * 32);  // next slab's loads fly during WMMAs
    __syncthreads();                      // lds_a[j&1] published to all waves

    Frag32 bnext[4];
    if (j + 1 < nk) loadB((j + 1) * 32, bnext);

    // A fragment: lane holds row l15; K = lhi*16..+15
    Frag32 afr[4];
#pragma unroll
    for (int i = 0; i < 4; i++) {
      const int rl = msub + i * 16 + l15;
      const uint4* a = (const uint4*)&lds_a[j & 1][rl * 32 + lhi * 16];
      afr[i].u[0] = a[0];
      afr[i].u[1] = a[1];
    }
#pragma unroll
    for (int i = 0; i < 4; i++)
#pragma unroll
      for (int jj = 0; jj < 4; jj++)
        acc[i][jj] = __builtin_amdgcn_wmma_f32_16x16x32_bf16(
            false, afr[i].v, false, bcur[jj].v, (short)0, acc[i][jj], false, false);

    if (j + 1 < nk) storeA((j + 1) & 1);  // tail store into the other buffer

#pragma unroll
    for (int jj = 0; jj < 4; jj++) bcur[jj] = bnext[jj];
  }

  // Epilogue: C/D layout — VGPR r: row = r + lhi*8, col = l15
#pragma unroll
  for (int j = 0; j < 4; j++) {
    const int n = ntile + nsub + j * 16 + l15;
    const float bv = bias ? bias[n] : 0.0f;
#pragma unroll
    for (int i = 0; i < 4; i++) {
      if (OUT_MODE == 0) {
#pragma unroll
        for (int r = 0; r < 8; r++) {
          const int m = mtile + msub + i * 16 + r + lhi * 8;
          ((float*)Cp)[(long)m * N + n] = acc[i][j][r] + bv;
        }
      } else if (n < 1536) {               // Q|K rows (uniform per 16-col tile)
#pragma unroll
        for (int r = 0; r < 8; r++) {
          const int m = mtile + msub + i * 16 + r + lhi * 8;
          ((unsigned short*)Cp)[(long)m * 1536 + n] = f2bf(acc[i][j][r] + bv);
        }
      } else {                             // V -> transposed, 8 keys per lane
        const int c  = n - 1536;
        const int hh = c / 192, dh = c % 192;
        const int m0 = mtile + msub + i * 16 + lhi * 8;  // 8 consecutive keys
        const int bb = m0 >> 11, key = m0 & 2047;        // L = 2048
        union { unsigned short s[8]; uint4 u; } pk;
#pragma unroll
        for (int r = 0; r < 8; r++) pk.s[r] = f2bf(acc[i][j][r] + bv);
        *(uint4*)&vtOut[(((long)bb * 4 + hh) * 192 + dh) * 2048 + key] = pk.u;
      }
    }
  }
}

// ---------------------------------------------------------------------------
// Kernel 3: flash attention (unscaled scores, per reference).
//   qk:  [B*L, 1536] bf16  (Q | K, head h at col h*192 within segment)
//   vt:  [B, H, 192, L] bf16 (V pre-transposed by the QKV GEMM epilogue)
//   ctx: [B*L, 768]  bf16
//   Block: 128 threads = 4 wave32; each wave owns 16 q rows; key blocks of 32.
//   K / V^T tiles double-buffered in LDS via global_load_async_to_lds_b128.
// ---------------------------------------------------------------------------
__global__ __launch_bounds__(128) void flash_attn(
    const unsigned short* __restrict__ qk, const unsigned short* __restrict__ vt,
    unsigned short* __restrict__ ctx, int L) {
  __shared__ __align__(16) unsigned short lds_k[2][32 * 192];  // [key][dh]   12KB x2
  __shared__ __align__(16) unsigned short lds_v[2][192 * 32];  // [dh][key]   12KB x2
  __shared__ __align__(16) unsigned short lds_p[4][512];       // per-wave probs

  const int tid  = threadIdx.x;
  const int lane = tid & 31;
  const int wave = tid >> 5;
  const int l15  = lane & 15;
  const int lhi  = lane >> 4;

  const int h = blockIdx.y;
  const int b = blockIdx.z;
  const long rowbase = (long)b * L;
  const int q0 = blockIdx.x * 64 + wave * 16;
  const int bh = b * 4 + h;

  const int QO = h * 192;
  const int KO = 768 + h * 192;

  // Issue one K+V tile's async copies (24 KB => 48 b128 instrs per block,
  // 12 per wave). j<6 covers the K tile, j>=6 the V^T tile (uniform branch).
  auto issue_tile = [&](int kb2, int buf) {
    const int key0n = kb2 * 32;
#pragma unroll
    for (int j = 0; j < 12; j++) {
      const int c = j * 128 + tid;  // 16B-chunk index in [0, 1536)
      if (c < 768) {
        const int key = c / 24, part = c % 24;  // K row = 384B = 24 chunks
        const unsigned short* g =
            qk + (rowbase + key0n + key) * 1536 + KO + part * 8;
        const unsigned loff =
            (unsigned)(uintptr_t)&lds_k[buf][key * 192 + part * 8];
        async_copy_b128(loff, g);
      } else {
        const int c2 = c - 768;
        const int dh = c2 / 4, part = c2 % 4;   // V^T row = 64B = 4 chunks
        const unsigned short* g =
            vt + ((long)bh * 192 + dh) * L + key0n + part * 8;
        const unsigned loff =
            (unsigned)(uintptr_t)&lds_v[buf][dh * 32 + part * 8];
        async_copy_b128(loff, g);
      }
    }
  };

  // Preload 6 Q fragments (A-matrix layout; reused across all key blocks)
  Frag32 qf[6];
  {
    const unsigned short* qrow = qk + (rowbase + q0 + l15) * 1536 + QO;
#pragma unroll
    for (int t = 0; t < 6; t++) {
      const uint4* p = (const uint4*)(qrow + t * 32 + lhi * 16);
      qf[t].u[0] = p[0];
      qf[t].u[1] = p[1];  // K = lhi*16 + 8 .. +15
    }
  }

  float mrow[8], lrow[8];
#pragma unroll
  for (int r = 0; r < 8; r++) { mrow[r] = -1e30f; lrow[r] = 0.0f; }
  v8f Oacc[12];
#pragma unroll
  for (int t = 0; t < 12; t++) Oacc[t] = zero8();

  const int nkb = L / 32;
  issue_tile(0, 0);

  for (int kb = 0; kb < nkb; ++kb) {
    const int buf = kb & 1;
    if (kb + 1 < nkb) {
      issue_tile(kb + 1, buf ^ 1);
      asm volatile("s_wait_asynccnt 0xc" ::: "memory");  // tile kb landed
    } else {
      asm volatile("s_wait_asynccnt 0x0" ::: "memory");
    }
    __syncthreads();

    // Scores: S[16q, 32keys] = Q @ K^T, two 16x16 tiles, 6 WMMAs each.
    // B fragment: lane = key col, 16 contiguous dh (row-major K tile in LDS).
    v8f s0 = zero8(), s1 = zero8();
#pragma unroll
    for (int t = 0; t < 6; t++) {
      Frag32 kf0, kf1;
      const uint4* kr0 = (const uint4*)&lds_k[buf][l15 * 192 + t * 32 + lhi * 16];
      const uint4* kr1 = (const uint4*)&lds_k[buf][(16 + l15) * 192 + t * 32 + lhi * 16];
      kf0.u[0] = kr0[0]; kf0.u[1] = kr0[1];
      kf1.u[0] = kr1[0]; kf1.u[1] = kr1[1];
      s0 = __builtin_amdgcn_wmma_f32_16x16x32_bf16(false, qf[t].v, false, kf0.v,
                                                   (short)0, s0, false, false);
      s1 = __builtin_amdgcn_wmma_f32_16x16x32_bf16(false, qf[t].v, false, kf1.v,
                                                   (short)0, s1, false, false);
    }

    // Online softmax: per-row (reg r -> row r + lhi*8) max/sum across 16-lane half
    float p0[8], p1[8], alpha[8];
#pragma unroll
    for (int r = 0; r < 8; r++) {
      float x = fmaxf(s0[r], s1[r]);
#pragma unroll
      for (int msk = 1; msk < 16; msk <<= 1) x = fmaxf(x, __shfl_xor(x, msk, 32));
      const float mn = fmaxf(mrow[r], x);
      alpha[r] = __expf(mrow[r] - mn);
      p0[r] = __expf(s0[r] - mn);
      p1[r] = __expf(s1[r] - mn);
      float sm = p0[r] + p1[r];
#pragma unroll
      for (int msk = 1; msk < 16; msk <<= 1) sm += __shfl_xor(sm, msk, 32);
      lrow[r] = lrow[r] * alpha[r] + sm;
      mrow[r] = mn;
    }
#pragma unroll
    for (int t = 0; t < 12; t++)
#pragma unroll
      for (int r = 0; r < 8; r++) Oacc[t][r] *= alpha[r];

    // Round-trip probs through wave-private LDS: C-layout -> A-fragment layout
    unsigned short* pw = lds_p[wave];
#pragma unroll
    for (int r = 0; r < 8; r++) {
      const int row = r + lhi * 8;
      pw[row * 32 + l15]      = f2bf(p0[r]);
      pw[row * 32 + 16 + l15] = f2bf(p1[r]);
    }
    asm volatile("s_wait_dscnt 0x0" ::: "memory");  // wave-internal LDS ordering
    Frag32 pf;
    {
      const uint4* p = (const uint4*)(pw + l15 * 32 + lhi * 16);
      pf.u[0] = p[0];
      pf.u[1] = p[1];
    }

    // O += P @ V : 12 dh tiles, B fragments from transposed V tile in LDS
#pragma unroll
    for (int t = 0; t < 12; t++) {
      Frag32 vf;
      const uint4* p = (const uint4*)&lds_v[buf][(t * 16 + l15) * 32 + lhi * 16];
      vf.u[0] = p[0];
      vf.u[1] = p[1];
      Oacc[t] = __builtin_amdgcn_wmma_f32_16x16x32_bf16(false, pf.v, false, vf.v,
                                                        (short)0, Oacc[t], false, false);
    }

    __syncthreads();  // all waves done reading buf before it is refilled
  }

  // Normalize and store ctx (bf16)
#pragma unroll
  for (int r = 0; r < 8; r++) lrow[r] = 1.0f / lrow[r];
#pragma unroll
  for (int t = 0; t < 12; t++) {
#pragma unroll
    for (int r = 0; r < 8; r++) {
      const int m = q0 + r + lhi * 8;
      ctx[(rowbase + m) * 768 + h * 192 + t * 16 + l15] = f2bf(Oacc[t][r] * lrow[r]);
    }
  }
}

// ---------------------------------------------------------------------------
// Host launcher
// ---------------------------------------------------------------------------
extern "C" void kernel_launch(void* const* d_in, const int* in_sizes, int n_in,
                              void* d_out, int out_size, void* d_ws, size_t ws_size,
                              hipStream_t stream) {
  const float* samples = (const float*)d_in[0];
  const float* Wq = (const float*)d_in[1];
  const float* bq = (const float*)d_in[2];
  const float* Wk = (const float*)d_in[3];
  const float* bk = (const float*)d_in[4];
  const float* Wv = (const float*)d_in[5];
  const float* bv = (const float*)d_in[6];
  const float* Wo = (const float*)d_in[7];
  const float* bo = (const float*)d_in[8];

  const int B = 8, L = 2048;
  const int M = in_sizes[0] / 1536;  // B*L = 16384

  // Workspace layout (all offsets 256B aligned):
  char* ws = (char*)d_ws;
  unsigned short* WcatT    = (unsigned short*)(ws + 0);          //  7,077,888 B
  unsigned short* WoT      = (unsigned short*)(ws + 7077888);    //  1,179,648 B
  float*          bias_cat = (float*)        (ws + 8257536);     //      9,216 B
  unsigned short* qk       = (unsigned short*)(ws + 8266752);    // 50,331,648 B
  unsigned short* vtbuf    = (unsigned short*)(ws + 58598400);   // 25,165,824 B
  unsigned short* ctx      = (unsigned short*)(ws + 83764224);   // 25,165,824 B

  pack_weights<<<2048, 256, 0, stream>>>(Wq, Wk, Wv, Wo, bq, bk, bv,
                                         WcatT, WoT, bias_cat);

  // Fused QKV projection: [M,1536] @ [1536,2304] -> bf16 qk + transposed V
  dim3 g1(2304 / 256, M / 128);
  gemm_bf16_wmma<0, 1><<<g1, 256, 0, stream>>>(samples, WcatT, bias_cat, qk,
                                               vtbuf, M, 2304, 1536);

  // Flash attention per (q-block of 64, head, batch)
  dim3 g2(L / 64, 4, B);
  flash_attn<<<g2, 128, 0, stream>>>(qk, vtbuf, ctx, L);

  // Output projection: [M,768] @ [768,768] + bo -> fp32 d_out
  dim3 g3(768 / 256, M / 128);
  gemm_bf16_wmma<1, 0><<<g3, 256, 0, stream>>>(ctx, WoT, bo, d_out, nullptr,
                                               M, 768, 768);
}
